// TwoDimensionalSSM_37993280700539
// MI455X (gfx1250) — compile-verified, matrix operand order checked by reading, not run on hardware
//
#include <hip/hip_runtime.h>
#include <hip/hip_bf16.h>
#include <stdint.h>

// ---- problem constants (match reference) ----
#define LZ   32
#define EDC  32
#define NDC  2
#define KDC  128
#define RR   65536
#define VV   64
#define HHI  64
#define BBI  8
#define HN   256          // KD*ND
#define SCALE_C 0.70710678118654752f

typedef __bf16 bf16_t;
typedef __attribute__((ext_vector_type(16))) __bf16 v16bf;
typedef __attribute__((ext_vector_type(8)))  float  v8f;

static __device__ inline v8f wmma_bf16(v16bf a, v16bf b, v8f c) {
  // D(16x16,f32) = A(16x32,bf16) x B(32x16,bf16) + C
  return __builtin_amdgcn_wmma_f32_16x16x32_bf16(false, a, false, b, (short)0, c, false, false);
}

// =====================================================================
// Kernel 1: build B-operands pre-swizzled into wave32 WMMA fragment
// order:  vecswz[m][nt][lane][j]  (j<16 -> K-frag0, j>=16 -> K-frag1)
// =====================================================================
__global__ void __launch_bounds__(256) vec_precompute(
    const float* __restrict__ A1, const float* __restrict__ A2,
    const float* __restrict__ A3, const float* __restrict__ A4,
    const float* __restrict__ B1, const float* __restrict__ B2,
    bf16_t* __restrict__ vecswz, float* __restrict__ BstkF)
{
  int tid = blockIdx.x * blockDim.x + threadIdx.x;
  const int total = 4 * 16 * 32 * 32;   // 65536
  for (int idx = tid; idx < total; idx += gridDim.x * blockDim.x) {
    int m    = idx >> 14;
    int nt   = (idx >> 10) & 15;
    int lane = (idx >> 5) & 31;
    int j    = idx & 31;
    int e    = j & 15;
    int k    = 32 * (j >> 4) + ((e < 8 ? e : e + 8) + (lane >> 4) * 8);
    int n    = nt * 16 + (lane & 15);
    const float* A = (m == 0) ? A1 : (m == 1) ? A2 : (m == 2) ? A3 : A4;
    float sig = 1.0f / (1.0f + __expf(-A[n]));
    float val = __expf((float)k * __logf(sig));   // sig^k
    vecswz[idx] = (bf16_t)val;
  }
  if (tid < 2 * HN) {
    int s = tid / HN, col = tid % HN;
    const float* Bp = (s == 0) ? B1 : B2;
    BstkF[tid] = 1.0f / (1.0f + __expf(-Bp[col]));
  }
}

// =====================================================================
// Kernel 2: one workgroup per l in [0,1024).  Register-resident 5-factor
// product via 8 chained WMMAs per tile, ds_add_f32 reduction.
// =====================================================================
__global__ void __launch_bounds__(256) ssm_kernel(
    const float* __restrict__ Mh1, const float* __restrict__ Mh2,
    const float* __restrict__ Mh3, const float* __restrict__ Mh4,
    const float* __restrict__ MhB,
    const float* __restrict__ Mv1, const float* __restrict__ Mv2,
    const float* __restrict__ Mv3, const float* __restrict__ Mv4,
    const float* __restrict__ MvB,
    const bf16_t* __restrict__ vecswz, const float* __restrict__ BstkF,
    const float* __restrict__ C1, const float* __restrict__ C2,
    float* __restrict__ kker)
{
  extern __shared__ char smem[];
  bf16_t* sM   = (bf16_t*)smem;                     // [8][64][72] bf16 = 73728 B
  float*  sMB  = (float*)(smem + 8 * 64 * 72 * 2);  // [2][128]
  float*  sVB  = sMB + 256;                         // [512] Bstk f32
  float*  Scol = sVB + 512;                         // [2][256] reduction target

  const int l    = blockIdx.x;
  const int tid  = threadIdx.x;
  const int lane = tid & 31;
  const int wv   = tid >> 5;
  const int khi  = (lane >> 4) * 8;
  const int l15  = lane & 15;

  if (tid < 512) Scol[tid] = 0.0f;
  for (int idx = tid; idx < 512; idx += 256) sVB[idx] = BstkF[idx];
  if (tid < 256) {
    const float* MB = (tid < 128) ? MhB : MvB;
    sMB[tid] = MB[(size_t)l * 128 + (tid & 127)];
  }
  // warm WGP$/L2 for the B-fragment table (-> global_prefetch_b8)
  __builtin_prefetch(vecswz + (size_t)tid * 256, 0, 1);

  // stage all 8 M matrices (both dirs) as bf16 A-operands
  for (int s = 0; s < 8; ++s) {
    const float* Mp =
        (s == 0) ? Mh1 : (s == 1) ? Mh2 : (s == 2) ? Mh3 : (s == 3) ? Mh4 :
        (s == 4) ? Mv1 : (s == 5) ? Mv2 : (s == 6) ? Mv3 : Mv4;
    Mp += (size_t)l * 64 * 64;
    bf16_t* dst = sM + s * (64 * 72);
    for (int idx = tid; idx < 4096; idx += 256)
      dst[(idx >> 6) * 72 + (idx & 63)] = (bf16_t)Mp[idx];
  }
  __syncthreads();

  for (int ntv = 0; ntv < 2; ++ntv) {
    const int nt   = wv + ntv * 8;
    const int ncol = nt * 16 + l15;

    // B fragments for all 4 vec matrices, held in registers across dirs
    v16bf bfr[4][2];
#pragma unroll
    for (int m = 0; m < 4; ++m) {
      const v16bf* p = (const v16bf*)(vecswz + (((size_t)(m * 16 + nt) * 32 + lane) * 32));
      bfr[m][0] = p[0];
      bfr[m][1] = p[1];
    }
    const float vb0 = sVB[ncol], vb1 = sVB[256 + ncol];

    for (int d = 0; d < 2; ++d) {
#pragma unroll
      for (int mt = 0; mt < 4; ++mt) {
        const int mrow = mt * 16 + l15;
        v8f prod = {1.f, 1.f, 1.f, 1.f, 1.f, 1.f, 1.f, 1.f};
#pragma unroll
        for (int m = 0; m < 4; ++m) {
          const bf16_t* ap = &sM[((d * 4 + m) * 64 + mrow) * 72 + khi];
          v16bf a0, a1;
#pragma unroll
          for (int e = 0; e < 8; ++e) {
            a0[e] = ap[e];      a0[8 + e] = ap[16 + e];
            a1[e] = ap[32 + e]; a1[8 + e] = ap[48 + e];
          }
          v8f acc = {};
          acc = wmma_bf16(a0, bfr[m][0], acc);
          acc = wmma_bf16(a1, bfr[m][1], acc);
          prod = prod * acc;           // elementwise, register resident
        }
        // fold in the M_B * Bstk factor and reduce the lane's 8 rows
        float partial = 0.0f;
#pragma unroll
        for (int v = 0; v < 8; ++v) {
          int g = mt * 16 + khi + v;
          float mb = sMB[d * 128 + g * 2] * vb0 + sMB[d * 128 + g * 2 + 1] * vb1;
          partial += prod[v] * mb;
        }
        atomicAdd(&Scol[d * 256 + ncol], partial);   // ds_add_f32
      }
    }
  }
  __syncthreads();

  // C contraction over (d, n), boundary scale, store k[h][l]
  if (tid < 128) {
    int h = tid;
    float v = Scol[2 * h]       * C1[2 * h] + Scol[2 * h + 1]       * C1[2 * h + 1]
            + Scol[256 + 2 * h] * C2[2 * h] + Scol[256 + 2 * h + 1] * C2[2 * h + 1];
    v *= SCALE_C;
    int i = l >> 5, j = l & 31;
    float f = ((i == 0) != (j == 0)) ? 2.0f : 1.0f;
    kker[(size_t)h * 1024 + l] = v * f;
  }
}

// =====================================================================
// Kernel 3: 4-direction depthwise conv. One workgroup per (b,c).
// stage1 (WMMA): for t-row mt, both a-tiles, write skewed by output row
//     i = t - si*a + s0   ->   Zskew[i][a][jj]  (two ds_store_b128)
// stage2 (VALU): y[i][j0+jj] += sum_a Zskew[i][a][jj]
// Strides: i-stride 672 (=32 mod 64), a-stride 20: stage2 is
// bank-conflict free; stores are 16B-aligned.
// =====================================================================
__global__ void __launch_bounds__(256) conv_kernel(
    const float* __restrict__ x, const float* __restrict__ kker,
    float* __restrict__ y)
{
  extern __shared__ char smem[];
  bf16_t* xp   = (bf16_t*)smem;                 // [128][136] zero-padded plane, bf16
  bf16_t* sK   = xp + 128 * 136;                // [32][40]  K[a][beta] bf16
  float*  Z    = (float*)(smem + (128 * 136 + 32 * 40) * 2); // skewed [64]x672
  float*  yacc = Z + 64 * 672;                  // [64][64] f32

  const int b = blockIdx.x >> 5;
  const int c = blockIdx.x & 31;
  const int tid  = threadIdx.x;
  const int lane = tid & 31;
  const int wv   = tid >> 5;
  const int khi  = (lane >> 4) * 8;
  const int l15  = lane & 15;

  for (int idx = tid; idx < 128 * 136; idx += 256) xp[idx] = (bf16_t)0.0f;
  for (int idx = tid; idx < 64 * 64;  idx += 256) yacc[idx] = 0.0f;
  __syncthreads();
  for (int idx = tid; idx < 4096; idx += 256) {
    int i = idx >> 6, j = idx & 63;
    xp[(i + 32) * 136 + (j + 32)] = (bf16_t)x[((size_t)(b * 64 + i) * 64 + j) * 32 + c];
  }
  __syncthreads();

  for (int d = 0; d < 4; ++d) {
    const int si = (d & 1) ? 1 : -1;
    const int sj = (d & 2) ? 1 : -1;
    const int s0 = (si == 1) ? 0 : -31;
    const int h  = d * 32 + c;
    for (int idx = tid; idx < 1024; idx += 256)
      sK[(idx >> 5) * 40 + (idx & 31)] = (bf16_t)kker[(size_t)h * 1024 + idx];
    __syncthreads();

    // hoist the two K-tile B-fragments (dir-dependent only): aligned b128s
    v16bf bk[2];
#pragma unroll
    for (int nt = 0; nt < 2; ++nt) {
      const bf16_t* kp = &sK[(nt * 16 + l15) * 40 + khi];
#pragma unroll
      for (int e = 0; e < 8; ++e) { bk[nt][e] = kp[e]; bk[nt][8 + e] = kp[16 + e]; }
    }

    for (int jt = 0; jt < 4; ++jt) {
      const int j0 = jt * 16;
      // ---- stage 1: 96 t-rows; one A-fragment feeds both a-tiles ----
      for (int mt = wv; mt < 96; mt += 8) {
        const int xr  = 32 + s0 + mt;
        const int xc0 = 32 + j0 + l15;
        const bf16_t* xrow = &xp[xr * 136];
        v16bf a;
#pragma unroll
        for (int e = 0; e < 8; ++e) {
          a[e]     = xrow[xc0 + sj * (khi + e)];
          a[8 + e] = xrow[xc0 + sj * (khi + 16 + e)];
        }
#pragma unroll
        for (int nt = 0; nt < 2; ++nt) {
          v8f acc = {};
          acc = wmma_bf16(a, bk[nt], acc);
          const int aa = nt * 16 + l15;                 // this lane's a-column
          const int i  = (si > 0) ? (mt - aa) : (mt + aa - 31);
          if (0 <= i && i < 64) {
            float* zb = &Z[i * 672 + aa * 20 + khi];    // 16B-aligned
            float4 lo, hi;
            lo.x = acc[0]; lo.y = acc[1]; lo.z = acc[2]; lo.w = acc[3];
            hi.x = acc[4]; hi.y = acc[5]; hi.z = acc[6]; hi.w = acc[7];
            *(float4*)(zb)     = lo;
            *(float4*)(zb + 4) = hi;
          }
        }
      }
      __syncthreads();
      // ---- stage 2: reduce over a (conflict-free scalar reads) ----
      for (int q = tid; q < 1024; q += 256) {
        int i = q >> 4, jj = q & 15;
        float s = 0.0f;
#pragma unroll
        for (int a = 0; a < 32; ++a) s += Z[i * 672 + a * 20 + jj];
        yacc[i * 64 + j0 + jj] += s;
      }
      __syncthreads();
    }
  }

  for (int idx = tid; idx < 4096; idx += 256) {
    int i = idx >> 6, j = idx & 63;
    y[((size_t)(b * 64 + i) * 64 + j) * 32 + c] = yacc[i * 64 + j];
  }
}

// =====================================================================
// Kernel 4: out = y @ W^T + bias.  M=32768, N=32, K=32.  128 rows / WG.
// =====================================================================
__global__ void __launch_bounds__(256) proj_kernel(
    const float* __restrict__ y, const float* __restrict__ W,
    const float* __restrict__ bias, float* __restrict__ out)
{
  __shared__ bf16_t sY[128][40];
  __shared__ bf16_t sW[32][40];   // sW[co][c] (K-contiguous for B frag)
  __shared__ float  sb[32];

  const int tid  = threadIdx.x;
  const int lane = tid & 31;
  const int wv   = tid >> 5;
  const int khi  = (lane >> 4) * 8;
  const int l15  = lane & 15;
  const int m0   = blockIdx.x * 128;

  for (int idx = tid; idx < 4096; idx += 256)
    sY[idx >> 5][idx & 31] = (bf16_t)y[(size_t)(m0 + (idx >> 5)) * 32 + (idx & 31)];
  for (int idx = tid; idx < 1024; idx += 256)
    sW[idx >> 5][idx & 31] = (bf16_t)W[idx];
  if (tid < 32) sb[tid] = bias[tid];
  __syncthreads();

  for (int q = wv * 2; q < wv * 2 + 2; ++q) {   // 16 jobs, 2 per wave
    int mt = q >> 1, nt = q & 1;
    int row = mt * 16 + l15;
    int co  = nt * 16 + l15;
    const bf16_t* ap = &sY[row][khi];
    const bf16_t* bp = &sW[co][khi];
    v16bf a, bb;
#pragma unroll
    for (int e = 0; e < 8; ++e) {
      a[e] = ap[e];   a[8 + e] = ap[16 + e];
      bb[e] = bp[e];  bb[8 + e] = bp[16 + e];
    }
    v8f acc = {};
    acc = wmma_bf16(a, bb, acc);
#pragma unroll
    for (int v = 0; v < 8; ++v) {
      int m = m0 + mt * 16 + khi + v;
      out[(size_t)m * 32 + co] = acc[v] + sb[co];
    }
  }
}

// =====================================================================
extern "C" void kernel_launch(void* const* d_in, const int* in_sizes, int n_in,
                              void* d_out, int out_size, void* d_ws, size_t ws_size,
                              hipStream_t stream) {
  (void)in_sizes; (void)n_in; (void)out_size; (void)ws_size;
  const float* x   = (const float*)d_in[0];
  const float* Mh1 = (const float*)d_in[1];
  const float* Mh2 = (const float*)d_in[2];
  const float* Mh3 = (const float*)d_in[3];
  const float* Mh4 = (const float*)d_in[4];
  const float* MhB = (const float*)d_in[5];
  const float* Mv1 = (const float*)d_in[6];
  const float* Mv2 = (const float*)d_in[7];
  const float* Mv3 = (const float*)d_in[8];
  const float* Mv4 = (const float*)d_in[9];
  const float* MvB = (const float*)d_in[10];
  const float* A1  = (const float*)d_in[11];
  const float* A2  = (const float*)d_in[12];
  const float* A3  = (const float*)d_in[13];
  const float* A4  = (const float*)d_in[14];
  const float* B1  = (const float*)d_in[15];
  const float* B2  = (const float*)d_in[16];
  const float* C1  = (const float*)d_in[17];
  const float* C2  = (const float*)d_in[18];
  const float* W   = (const float*)d_in[19];
  const float* bia = (const float*)d_in[20];
  float* out = (float*)d_out;

  char* ws = (char*)d_ws;
  bf16_t* vecswz = (bf16_t*)ws;                      // 65536 bf16 = 128 KB
  float*  BstkF  = (float*)(ws + 131072);            // 512 f32    = 2 KB
  float*  kker   = (float*)(ws + 133120);            // 128*1024 f32 = 512 KB
  float*  ybuf   = (float*)(ws + 133120 + 524288);   // 8*64*64*32 f32 = 4 MB

  vec_precompute<<<64, 256, 0, stream>>>(A1, A2, A3, A4, B1, B2, vecswz, BstkF);

  size_t ssm_lds = (size_t)(8 * 64 * 72) * 2                 // sM (bf16)
                 + (size_t)(256 + 512 + 512) * 4;            // sMB, sVB, Scol
  ssm_kernel<<<1024, 256, ssm_lds, stream>>>(Mh1, Mh2, Mh3, Mh4, MhB,
                                             Mv1, Mv2, Mv3, Mv4, MvB,
                                             vecswz, BstkF, C1, C2, kker);

  size_t conv_lds = (size_t)(128 * 136 + 32 * 40) * 2        // xp + sK (bf16)
                  + (size_t)(64 * 672 + 64 * 64) * 4;        // Zskew + yacc
  conv_kernel<<<256, 256, conv_lds, stream>>>(x, kker, ybuf);

  proj_kernel<<<256, 256, 0, stream>>>(ybuf, W, bia, out);
}
